// DynamicConv2d_43894565765572
// MI455X (gfx1250) — matrix operand order repaired and checked by reading, not compiled, not run on hardware
//
#include <hip/hip_runtime.h>
#include <math.h>

typedef __attribute__((ext_vector_type(16))) __bf16 v16bf;
typedef __attribute__((ext_vector_type(8)))  __bf16 v8bf;
typedef __attribute__((ext_vector_type(8)))  float  v8f;
typedef float f32x4u __attribute__((ext_vector_type(4), aligned(4)));

#define BATCH  32
#define CIN    256
#define COUT   256
#define HWDIM  56
#define NPIX   3136   // 56*56
#define KFLAT  2304   // 256*9
#define NBANK  4
#define HIDDEN 64

// ---------------------------------------------------------------------------
// 0) Zero the pad region used to source out-of-image rows
// ---------------------------------------------------------------------------
__global__ void zinit_kernel(float* __restrict__ z) {
    z[threadIdx.x] = 0.f;   // 16 floats
}

// ---------------------------------------------------------------------------
// 1) Global average pool: v[b*Cin + c] = mean(x[b,c,:,:])
// ---------------------------------------------------------------------------
__global__ __launch_bounds__(256)
void gap_kernel(const float* __restrict__ x, float* __restrict__ v) {
    __shared__ float red[256];
    const int bc = blockIdx.x;                  // b*Cin + c
    const float* p = x + (size_t)bc * NPIX;
    float s = 0.f;
    for (int i = threadIdx.x; i < NPIX; i += 256) s += p[i];
    red[threadIdx.x] = s;
    __syncthreads();
    for (int off = 128; off > 0; off >>= 1) {
        if (threadIdx.x < off) red[threadIdx.x] += red[threadIdx.x + off];
        __syncthreads();
    }
    if (threadIdx.x == 0) v[bc] = red[0] * (1.0f / (float)NPIX);
}

// ---------------------------------------------------------------------------
// 2) Router MLP + softmax: a[b, 0..3]
// ---------------------------------------------------------------------------
__global__ __launch_bounds__(HIDDEN)
void router_kernel(const float* __restrict__ v,
                   const float* __restrict__ fc1w, const float* __restrict__ fc1b,
                   const float* __restrict__ fc2w, const float* __restrict__ fc2b,
                   float* __restrict__ a) {
    __shared__ float hs[HIDDEN];
    __shared__ float lg[NBANK];
    const int b = blockIdx.x;
    const int h = threadIdx.x;                  // 64 threads
    const float* vb = v + b * CIN;
    float s = fc1b[h];
    for (int c = 0; c < CIN; ++c) s += vb[c] * fc1w[h * CIN + c];
    hs[h] = fmaxf(s, 0.f);
    __syncthreads();
    if (h < NBANK) {
        float l = fc2b[h];
        for (int j = 0; j < HIDDEN; ++j) l += hs[j] * fc2w[h * HIDDEN + j];
        lg[h] = l;                              // TEMPERATURE == 1.0
    }
    __syncthreads();
    if (h == 0) {
        float m = fmaxf(fmaxf(lg[0], lg[1]), fmaxf(lg[2], lg[3]));
        float e0 = __expf(lg[0] - m), e1 = __expf(lg[1] - m);
        float e2 = __expf(lg[2] - m), e3 = __expf(lg[3] - m);
        float inv = 1.0f / (e0 + e1 + e2 + e3);
        a[b * NBANK + 0] = e0 * inv;
        a[b * NBANK + 1] = e1 * inv;
        a[b * NBANK + 2] = e2 * inv;
        a[b * NBANK + 3] = e3 * inv;
    }
}

// ---------------------------------------------------------------------------
// 3) Mix the K banks into per-sample bf16 weights: wdyn[b][cout][f]
// ---------------------------------------------------------------------------
__global__ __launch_bounds__(256)
void mix_kernel(const float* __restrict__ bank, const float* __restrict__ a,
                __bf16* __restrict__ wdyn) {
    const size_t idx = (size_t)blockIdx.x * 256 + threadIdx.x; // [b][cout][f]
    const int f  = (int)(idx % KFLAT);
    const int bc = (int)(idx / KFLAT);
    const int cout = bc % COUT;
    const int b    = bc / COUT;
    const float* ap = a + b * NBANK;
    const size_t base = (size_t)cout * KFLAT + f;
    const size_t bs   = (size_t)COUT * KFLAT;      // stride between banks
    float s = ap[0] * bank[base]
            + ap[1] * bank[base + bs]
            + ap[2] * bank[base + 2 * bs]
            + ap[3] * bank[base + 3 * bs];
    wdyn[idx] = (__bf16)s;
}

// ---------------------------------------------------------------------------
// 4) Implicit-GEMM dynamic conv, bf16 WMMA (f32 accumulate).
//    Per sample b: Y[256,3136] = Wdyn_b[256,2304] x im2col(x_b)[2304,3136]
//    Block tile: M=256, N=64, K-step=32, double-buffered LDS B tile.
//    8 waves; wave w owns rows [32w, 32w+32) as two 16-row strips -> 8 WMMA/iter.
//    Gather: thread owns one K-row (k=tid>>3) and 8 consecutive pixels
//    ((tid&7)*8 .. +8), which never cross an image row since 8 | 56.
// ---------------------------------------------------------------------------
__global__ __launch_bounds__(256)
void dynconv_kernel(const float* __restrict__ x, const __bf16* __restrict__ wdyn,
                    const float* __restrict__ zeropad, float* __restrict__ y) {
    // B tile in WMMA B-fragment layout: [ngroup(4)][lane(32)][16 bf16], x2 buffers
    __shared__ __align__(32) __bf16 ldsB[2][4 * 32 * 16];

    const int bx   = blockIdx.x;
    const int nt   = bx % 49;            // 49 * 64 = 3136 pixels
    const int b    = bx / 49;
    const int n0   = nt * 64;
    const int tid  = threadIdx.x;
    const int lane = tid & 31;
    const int wave = tid >> 5;

    // ---- gather-role constants (hoisted) ----
    const int kIdx = tid >> 3;           // K-row within chunk: 0..31
    const int oct  = tid & 7;            // pixel octet: 0..7
    const int p0   = n0 + oct * 8;
    const int oh   = p0 / HWDIM;
    const int ow0  = p0 - oh * HWDIM;    // 8-aligned, octet stays in this row
    // incremental decomposition of kk = kc + kIdx : ci*NPIX and rem = kk % 9
    int rem = kIdx % 9;
    int ciN = (kIdx / 9) * NPIX;
    const float* xb = x + (size_t)b * CIN * NPIX;
    // LDS scatter addressing: group = oct>>1, lane_s = (oct&1)*8 + c + (kIdx>>4)*16,
    // element j = kIdx & 15; address step over c is 16 elements (32B)
    const int ldsWrBase = ((((oct >> 1) * 32) + ((oct & 1) << 3) + ((kIdx >> 4) << 4)) << 4)
                          + (kIdx & 15);

    // ---- compute-role constants ----
    const int mA   = wave * 32;          // first of two 16-row strips
    const int ksel = (lane >> 4) << 4;   // 0 or 16 (contiguous k = 16h + s)
    const __bf16* wrowA = wdyn + ((size_t)b * COUT + (mA + (lane & 15))) * KFLAT;
    const __bf16* wrowB = wrowA + (size_t)16 * KFLAT;

    v8f acc[2][4] = {};

    for (int kc = 0; kc < KFLAT; kc += 32) {
        __bf16* buf = ldsB[(kc >> 5) & 1];

        // ---- gather: one contiguous 8-float run per thread ----
        const int kh  = (rem * 11) >> 5;       // rem/3 for rem<9
        const int kw  = rem - 3 * kh;
        const int ih  = oh + kh - 1;           // PADDING = 1
        const int iwb = ow0 + kw - 1;
        const bool rowOK = (unsigned)ih < (unsigned)HWDIM;
        const float* src = rowOK ? (xb + ciN + ih * HWDIM + iwb) : zeropad;
        f32x4u lo = *(const f32x4u*)(src);
        f32x4u hi = *(const f32x4u*)(src + 4);
        lo.x = (iwb < 0)            ? 0.f : lo.x;   // left halo (only elem 0)
        hi.w = (iwb + 7 >= HWDIM)   ? 0.f : hi.w;   // right halo (only elem 7)
        float vals[8] = { lo.x, lo.y, lo.z, lo.w, hi.x, hi.y, hi.z, hi.w };
#pragma unroll
        for (int c = 0; c < 8; ++c)
            buf[ldsWrBase + (c << 4)] = (__bf16)vals[c];

        // advance kk decomposition: kk += 32 -> rem += 5 (mod 9), ci += 3 or 4
        rem += 5;
        if (rem >= 9) { rem -= 9; ciN += 4 * NPIX; } else { ciN += 3 * NPIX; }

        __syncthreads();

        // ---- compute: 2 A fragments (global, 16B-aligned), 4 B frags, 8 WMMAs ----
        union { v16bf v; v8bf h[2]; } aA, aB;
        aA.h[0] = *(const v8bf*)(wrowA + kc + ksel);
        aA.h[1] = *(const v8bf*)(wrowA + kc + ksel + 8);
        aB.h[0] = *(const v8bf*)(wrowB + kc + ksel);
        aB.h[1] = *(const v8bf*)(wrowB + kc + ksel + 8);
#pragma unroll
        for (int t = 0; t < 4; ++t) {
            v16bf bfrag = *(const v16bf*)(&buf[(t * 32 + lane) << 4]);
            acc[0][t] = __builtin_amdgcn_wmma_f32_16x16x32_bf16(
                            false, aA.v, false, bfrag, (short)0, acc[0][t], false, false);
            acc[1][t] = __builtin_amdgcn_wmma_f32_16x16x32_bf16(
                            false, aB.v, false, bfrag, (short)0, acc[1][t], false, false);
        }
        // single barrier per iteration: next gather targets the other buffer;
        // the barrier's dscnt wait makes the 2-iteration reuse safe.
    }

    // ---- store: lane<16 rows +0..7, lane>=16 rows +8..15 of each strip ----
    const int rbase = mA + ((lane >> 4) << 3);
    const int cbase = n0 + (lane & 15);
#pragma unroll
    for (int s = 0; s < 2; ++s) {
#pragma unroll
        for (int r = 0; r < 8; ++r) {
            float* yrow = y + ((size_t)b * COUT + (rbase + s * 16 + r)) * NPIX + cbase;
#pragma unroll
            for (int t = 0; t < 4; ++t)
                yrow[t * 16] = acc[s][t][r];
        }
    }
}

// ---------------------------------------------------------------------------
// Launch
// ---------------------------------------------------------------------------
extern "C" void kernel_launch(void* const* d_in, const int* in_sizes, int n_in,
                              void* d_out, int out_size, void* d_ws, size_t ws_size,
                              hipStream_t stream) {
    const float* x    = (const float*)d_in[0];
    const float* bank = (const float*)d_in[1];
    const float* fc1w = (const float*)d_in[2];
    const float* fc1b = (const float*)d_in[3];
    const float* fc2w = (const float*)d_in[4];
    const float* fc2b = (const float*)d_in[5];
    float* y = (float*)d_out;

    // ws layout: [ wdyn bf16 ][ v f32 32*256 ][ a f32 32*4 ][ zeropad f32 16 ]
    __bf16* wdyn = (__bf16*)d_ws;
    const size_t wdyn_bytes = (size_t)BATCH * COUT * KFLAT * sizeof(__bf16);
    float* v  = (float*)((char*)d_ws + wdyn_bytes);
    float* a  = v + BATCH * CIN;
    float* zp = a + BATCH * NBANK;

    zinit_kernel <<<1, 16, 0, stream>>>(zp);
    gap_kernel   <<<BATCH * CIN, 256, 0, stream>>>(x, v);
    router_kernel<<<BATCH, HIDDEN, 0, stream>>>(v, fc1w, fc1b, fc2w, fc2b, a);
    mix_kernel   <<<(BATCH * COUT * KFLAT) / 256, 256, 0, stream>>>(bank, a, wdyn);
    dynconv_kernel<<<BATCH * 49, 256, 0, stream>>>(x, wdyn, zp, y);
}
